// ConvEncoder_30923764531881
// MI455X (gfx1250) — matrix operand order repaired and checked by reading, not compiled, not run on hardware
//
#include <hip/hip_runtime.h>
#include <hip/hip_bf16.h>

typedef __attribute__((ext_vector_type(16))) _Float16 v16h;
typedef __attribute__((ext_vector_type(8)))  _Float16 v8h;
typedef __attribute__((ext_vector_type(8)))  float    v8f;
typedef __attribute__((ext_vector_type(4)))  float    v4f;

#define WMMA_F16(a, b, c) \
  __builtin_amdgcn_wmma_f32_16x16x32_f16(false, (a), false, (b), (short)0, (c), false, false)

__device__ __forceinline__ float sigf(float x) { return 1.f / (1.f + __expf(-x)); }

// Compose an A fragment (16x32 f16, contiguous k order) from two aligned 16B runs.
// p points at row-base + kb + kA0; runs are [0..7] and [8..15].
__device__ __forceinline__ v16h frag_a(const _Float16* p) {
  v8h lo = *(const v8h*)p;
  v8h hi = *(const v8h*)(p + 8);
  v16h a;
#pragma unroll
  for (int j = 0; j < 8; ++j) { a[j] = lo[j]; a[j + 8] = hi[j]; }
  return a;
}

// ---------------------------------------------------------------- fills / converts
__global__ void k_fill_u32(unsigned* p, unsigned v, long n) {
  long i = blockIdx.x * (long)blockDim.x + threadIdx.x;
  if (i < n) p[i] = v;
}

// src [N,K] row-major -> dst [K,N] f16 (transpose-convert). Used to build B^T [Nout,Kin].
__global__ void k_cvt_t(_Float16* dst, const float* src, int N, int K) {
  long i = blockIdx.x * (long)blockDim.x + threadIdx.x;
  if (i >= (long)N * K) return;
  int k = (int)(i / N), nn = (int)(i % N);
  dst[(long)k * N + nn] = (_Float16)src[(long)nn * K + k];
}

// dst[n*dstride + doff + k] = src[n*cols + k]  (strided row copy-convert)
__global__ void k_cvt_stride(_Float16* dst, const float* src, int rows, int cols,
                             int dstride, int doff) {
  long i = blockIdx.x * (long)blockDim.x + threadIdx.x;
  if (i >= (long)rows * cols) return;
  int n = (int)(i / cols), k = (int)(i % cols);
  dst[(long)n * dstride + doff + k] = (_Float16)src[i];
}

// dst[i,:] = emb[idx[i],:]
__global__ void k_gather_rows(float* dst, const float* emb, const int* idx, int M, int Dd) {
  long i = blockIdx.x * (long)blockDim.x + threadIdx.x;
  if (i >= (long)M * Dd) return;
  int r = (int)(i / Dd), d = (int)(i % Dd);
  dst[i] = emb[(long)idx[r] * Dd + d];
}

// ---------------------------------------------------------------- generic WMMA GEMM
// C[M,N] = gatherA(A)[M,K] @ B[K,N] (+bias), with B passed TRANSPOSED as Bt[N,K] f16.
// 256 thr = 8 waves; block tile 16 x 128; A panel staged once to LDS (K <= 256).
template <typename TA>
__global__ __launch_bounds__(256) void k_gemm(const TA* __restrict__ A,
                                              const int* __restrict__ rowidx,
                                              const _Float16* __restrict__ Bt,
                                              const float* __restrict__ bias,
                                              float* __restrict__ C, int M, int N, int K) {
  __shared__ _Float16 As[16 * 256];
  int tid = threadIdx.x;
  int mbase = blockIdx.x * 16;
  int k8n = K >> 3;                       // 8-element chunks per row
  for (int c = tid; c < 16 * k8n; c += 256) {
    int r = c / k8n, k8 = (c % k8n) * 8;
    int arow = mbase + r;
    long srow = (arow < M) ? (rowidx ? (long)rowidx[arow] : (long)arow) : 0;
    const TA* src = A + srow * K + k8;
#pragma unroll
    for (int j = 0; j < 8; ++j) As[r * K + k8 + j] = (_Float16)src[j];
  }
  __syncthreads();
  int wave = tid >> 5, lane = tid & 31;
  int nbase = blockIdx.y * 128 + wave * 16;
  if (nbase >= N) return;
  int arowL = lane & 15;
  int kA0 = (lane < 16) ? 0 : 16;
  int col = nbase + (lane & 15);
  int kB0 = (lane < 16) ? 0 : 16;
  const _Float16* Bcol = Bt + (long)col * K;
  v8f acc = {};
  for (int kb = 0; kb < K; kb += 32) {
    v16h a = frag_a(&As[arowL * K + kb + kA0]);
    v16h b = *(const v16h*)(Bcol + kb + kB0);   // 16 contiguous halfs (32B)
    acc = WMMA_F16(a, b, acc);
  }
  float bv = bias ? bias[col] : 0.f;
  int rbase = mbase + ((lane < 16) ? 0 : 8);
#pragma unroll
  for (int v = 0; v < 8; ++v) {
    int r = rbase + v;
    if (r < M) C[(long)r * N + col] = acc[v] + bv;
  }
}

// ---------------------------------------------------------------- BiLSTM recurrence
// grid.x = dir, grid.y = batch half (16 rows). Wt = f16 [512 gates][256 = xK|hK] (B^T layout).
__global__ __launch_bounds__(256) void k_lstm(const float* __restrict__ x,
                                              const float* __restrict__ mask,
                                              const _Float16* __restrict__ Wtf,
                                              const _Float16* __restrict__ Wtb,
                                              const float* __restrict__ bihf, const float* __restrict__ bhhf,
                                              const float* __restrict__ bihb, const float* __restrict__ bhhb,
                                              _Float16* __restrict__ hs_cat) {
  __shared__ _Float16 xh[16 * 256];   // [xt | h]  (A matrix, K=256)
  __shared__ float gates[16 * 512];
  __shared__ float cst[16 * 128];
  const int T = 160;
  int dir = blockIdx.x, b0 = blockIdx.y * 16, tid = threadIdx.x;
  const _Float16* Wt = dir ? Wtb : Wtf;
  const float* bih = dir ? bihb : bihf;
  const float* bhh = dir ? bhhb : bhhf;
  for (int i = tid; i < 16 * 128; i += 256) {
    int r = i >> 7, d = i & 127;
    xh[r * 256 + 128 + d] = (_Float16)0.f;
    cst[i] = 0.f;
  }
  int wave = tid >> 5, lane = tid & 31;
  int arowL = lane & 15, kA0 = (lane < 16) ? 0 : 16;
  int colL = lane & 15, kB0 = (lane < 16) ? 0 : 16;
  for (int t = 0; t < T; ++t) {
    int tt = dir ? (T - 1 - t) : t;
    __syncthreads();
    for (int c = tid; c < 16 * 32; c += 256) {    // stage x_t (vectorized f32x4)
      int r = c >> 5, d4 = (c & 31) * 4;
      v4f xv = *(const v4f*)&x[((long)(b0 + r) * T + tt) * 128 + d4];
#pragma unroll
      for (int j = 0; j < 4; ++j) xh[r * 256 + d4 + j] = (_Float16)xv[j];
    }
    __syncthreads();
    for (int nt = wave; nt < 32; nt += 8) {       // gates = [x,h] @ W  (16x512, K=256)
      int nb = nt * 16;
      const _Float16* Bcol = Wt + (long)(nb + colL) * 256;
      v8f acc = {};
      for (int kb = 0; kb < 256; kb += 32) {
        v16h a = frag_a(&xh[arowL * 256 + kb + kA0]);
        v16h b = *(const v16h*)(Bcol + kb + kB0);
        acc = WMMA_F16(a, b, acc);
      }
      int rb = (lane < 16) ? 0 : 8;
#pragma unroll
      for (int v = 0; v < 8; ++v)
        gates[(rb + v) * 512 + nb + colL] = acc[v] + bih[nb + colL] + bhh[nb + colL];
    }
    __syncthreads();
    for (int i = tid; i < 16 * 128; i += 256) {   // gate nonlinearity
      int r = i >> 7, d = i & 127;
      float gi = gates[r * 512 + d], gf = gates[r * 512 + 128 + d];
      float gg = gates[r * 512 + 256 + d], go = gates[r * 512 + 384 + d];
      float c = sigf(gf) * cst[i] + sigf(gi) * tanhf(gg);
      float h = sigf(go) * tanhf(c);
      cst[i] = c;
      xh[r * 256 + 128 + d] = (_Float16)h;
      float mk = mask[(long)(b0 + r) * T + tt];
      hs_cat[((long)(b0 + r) * T + tt) * 256 + dir * 128 + d] = (_Float16)(h * mk);
    }
  }
}

// ---------------------------------------------------------------- attention precompute
// att_edge[pos,h] = dot( (ws_table[pos] @ Wfeat + bfeat)[h*32:(h+1)*32], Wattn[h,32:64] )
__global__ void k_att_edge(float* tbl, const float* ws_table, const float* Wfeat,
                           const float* bfeat, const float* Wattn) {
  int i = blockIdx.x * blockDim.x + threadIdx.x;
  if (i >= 50 * 4) return;
  int pos = i >> 2, h = i & 3;
  float s = 0.f;
  for (int d = 0; d < 32; ++d) {
    float v = bfeat[h * 32 + d];
    for (int e = 0; e < 64; ++e) v += ws_table[pos * 64 + e] * Wfeat[e * 128 + h * 32 + d];
    s += v * Wattn[h * 64 + 32 + d];
  }
  tbl[i] = s;
}

// att_src[i,h] = dot(z[i, h*32:(h+1)*32], Wattn[h,0:32])
__global__ void k_att_src(float* att, const float* z, const float* Wattn, int Nsrc) {
  long i = blockIdx.x * (long)blockDim.x + threadIdx.x;
  if (i >= (long)Nsrc * 4) return;
  int r = (int)(i >> 2), h = (int)(i & 3);
  const float* zr = z + (long)r * 128 + h * 32;
  const float* w = Wattn + h * 64;
  float s = 0.f;
#pragma unroll
  for (int d = 0; d < 32; ++d) s += zr[d] * w[d];
  att[i] = s;
}

// ---------------------------------------------------------------- edge softmax passes
__global__ void k_edge_logits(float* ebuf, unsigned* m, const float* att_src,
                              const float* att_edge, const int* srcv, const int* dstv,
                              const int* link, long ne) {
  long i = blockIdx.x * (long)blockDim.x + threadIdx.x;
  if (i >= ne * 4) return;
  long n = i >> 2; int h = (int)(i & 3);
  float e = att_src[(long)srcv[n] * 4 + h] + att_edge[link[n] * 4 + h];
  e = e > 0.f ? e : 0.2f * e;                       // leaky_relu(0.2)
  ebuf[i] = e;
  unsigned u = __float_as_uint(e);
  unsigned key = (u & 0x80000000u) ? ~u : (u | 0x80000000u);   // order-preserving map
  atomicMax(&m[(long)dstv[n] * 4 + h], key);
}

__global__ void k_edge_expsum(float* ebuf, const unsigned* m, float* den,
                              const int* dstv, long ne) {
  long i = blockIdx.x * (long)blockDim.x + threadIdx.x;
  if (i >= ne * 4) return;
  long n = i >> 2; int h = (int)(i & 3);
  unsigned key = m[(long)dstv[n] * 4 + h];
  unsigned u = (key & 0x80000000u) ? (key & 0x7fffffffu) : ~key;
  float ex = __expf(ebuf[i] - __uint_as_float(u));
  ebuf[i] = ex;
  atomicAdd(&den[(long)dstv[n] * 4 + h], ex);
}

__global__ void k_edge_scatter(float* out, const float* ebuf, const float* den,
                               const float* z, const int* srcv, const int* dstv, long ne) {
  long i = blockIdx.x * (long)blockDim.x + threadIdx.x;
  if (i >= ne * 128) return;
  long n = i >> 7; int d = (int)(i & 127); int h = d >> 5;
  int dn = dstv[n];
  float dv = den[(long)dn * 4 + h];
  float alpha = ebuf[n * 4 + h] / (dv == 0.f ? 1.f : dv);
  atomicAdd(&out[(long)dn * 128 + d], alpha * z[(long)srcv[n] * 128 + d]);
}

// ---------------------------------------------------------------- fused elu+res+LN+FFN
// dest = h + relu(LN(h) @ W1 + b1) @ W2 + b2, where h = elu(accum) + origin. 16 rows/block.
// W1t f16 [512,128] = W1^T ; W2t f16 [128,512] = W2^T (B^T layout for contiguous frags).
__global__ __launch_bounds__(128) void k_post_ffn(const float* __restrict__ accum,
                                                  const float* __restrict__ origin,
                                                  float* __restrict__ dest,
                                                  const _Float16* __restrict__ W1t, const float* __restrict__ b1,
                                                  const _Float16* __restrict__ W2t, const float* __restrict__ b2,
                                                  const float* __restrict__ gamma, const float* __restrict__ beta,
                                                  int M) {
  __shared__ float hbuf[16 * 128];
  __shared__ _Float16 lnh[16 * 128];
  __shared__ _Float16 inner[16 * 512];
  __shared__ float stats[16 * 2];
  int tid = threadIdx.x;
  int mbase = blockIdx.x * 16;
  for (int i = tid; i < 16 * 128; i += 128) {
    int r = i >> 7, d = i & 127;
    long gr = mbase + r;
    float a = 0.f, o = 0.f;
    if (gr < M) { a = accum[gr * 128 + d]; o = origin[gr * 128 + d]; }
    float e = a > 0.f ? a : (__expf(a) - 1.f);     // elu
    hbuf[i] = e + o;
  }
  __syncthreads();
  {                                                 // LayerNorm stats, 8 lanes per row
    int r = tid >> 3, s = tid & 7;
    float sum = 0.f;
    for (int d = s * 16; d < s * 16 + 16; ++d) sum += hbuf[r * 128 + d];
    sum += __shfl_xor(sum, 1); sum += __shfl_xor(sum, 2); sum += __shfl_xor(sum, 4);
    float mu = sum * (1.f / 128.f);
    float vs = 0.f;
    for (int d = s * 16; d < s * 16 + 16; ++d) { float df = hbuf[r * 128 + d] - mu; vs += df * df; }
    vs += __shfl_xor(vs, 1); vs += __shfl_xor(vs, 2); vs += __shfl_xor(vs, 4);
    if (s == 0) { stats[r * 2] = mu; stats[r * 2 + 1] = rsqrtf(vs * (1.f / 128.f) + 1e-5f); }
  }
  __syncthreads();
  for (int i = tid; i < 16 * 128; i += 128) {
    int r = i >> 7, d = i & 127;
    lnh[i] = (_Float16)((hbuf[i] - stats[r * 2]) * stats[r * 2 + 1] * gamma[d] + beta[d]);
  }
  __syncthreads();
  int wave = tid >> 5, lane = tid & 31;
  int arowL = lane & 15, kA0 = (lane < 16) ? 0 : 16;
  int colL = lane & 15, kB0 = (lane < 16) ? 0 : 16;
  for (int nt = wave; nt < 32; nt += 4) {           // inner = relu(LN(h)@W1+b1) [16,512]
    int nb = nt * 16;
    const _Float16* Bcol = W1t + (long)(nb + colL) * 128;
    v8f acc = {};
    for (int kb = 0; kb < 128; kb += 32) {
      v16h a = frag_a(&lnh[arowL * 128 + kb + kA0]);
      v16h b = *(const v16h*)(Bcol + kb + kB0);
      acc = WMMA_F16(a, b, acc);
    }
    int rb = (lane < 16) ? 0 : 8;
#pragma unroll
    for (int v = 0; v < 8; ++v) {
      float val = acc[v] + b1[nb + colL];
      inner[(rb + v) * 512 + nb + colL] = (_Float16)(val > 0.f ? val : 0.f);
    }
  }
  __syncthreads();
  for (int nt = wave; nt < 8; nt += 4) {            // dest = inner@W2 + b2 + h [16,128]
    int nb = nt * 16;
    const _Float16* Bcol = W2t + (long)(nb + colL) * 512;
    v8f acc = {};
    for (int kb = 0; kb < 512; kb += 32) {
      v16h a = frag_a(&inner[arowL * 512 + kb + kA0]);
      v16h b = *(const v16h*)(Bcol + kb + kB0);
      acc = WMMA_F16(a, b, acc);
    }
    int rb = (lane < 16) ? 0 : 8;
#pragma unroll
    for (int v = 0; v < 8; ++v) {
      long gr = mbase + rb + v;
      if (gr < M) dest[gr * 128 + nb + colL] = acc[v] + b2[nb + colL] + hbuf[(rb + v) * 128 + nb + colL];
    }
  }
}

// ================================================================ host orchestration
static inline long cdivl(long a, long b) { return (a + b - 1) / b; }

extern "C" void kernel_launch(void* const* d_in, const int* in_sizes, int n_in,
                              void* d_out, int out_size, void* d_ws, size_t ws_size,
                              hipStream_t stream) {
  (void)n_in; (void)out_size; (void)ws_size;
  const int NW = in_sizes[2], NS = in_sizes[3];
  const long NE = in_sizes[5];
  const int BT = in_sizes[0] / 128;   // 5120

  // ---- inputs
  const float* local_feature = (const float*)d_in[0];
  const float* conv_mask     = (const float*)d_in[1];
  const int*   wid        = (const int*)d_in[2];
  const int*   ner_id     = (const int*)d_in[3];
  const int*   utter_idx  = (const int*)d_in[4];
  const int*   edge_src   = (const int*)d_in[5];
  const int*   edge_dst   = (const int*)d_in[6];
  const int*   ws_link    = (const int*)d_in[7];
  const float* word_emb   = (const float*)d_in[8];
  const float* ner_emb    = (const float*)d_in[9];
  const float* ws_table   = (const float*)d_in[10];
  const float* W_ner_proj = (const float*)d_in[11];
  const float* W_sent_proj= (const float*)d_in[12];
  const float* Wih_f = (const float*)d_in[13]; const float* Whh_f = (const float*)d_in[14];
  const float* bih_f = (const float*)d_in[15]; const float* bhh_f = (const float*)d_in[16];
  const float* Wih_b = (const float*)d_in[17]; const float* Whh_b = (const float*)d_in[18];
  const float* bih_b = (const float*)d_in[19]; const float* bhh_b = (const float*)d_in[20];
  // GAT param sets: [0]=w2s (d_in 21..30), [1]=s2w (d_in 31..40)
  const float *Wfc[2], *Wfeat[2], *bfeat[2], *Wattn[2], *W1[2], *b1[2], *W2[2], *b2[2], *gam[2], *bet[2];
  for (int s = 0; s < 2; ++s) {
    int base = 21 + 10 * s;
    Wfc[s]   = (const float*)d_in[base + 0];
    Wfeat[s] = (const float*)d_in[base + 1];
    bfeat[s] = (const float*)d_in[base + 2];
    Wattn[s] = (const float*)d_in[base + 3];
    W1[s]    = (const float*)d_in[base + 4];
    b1[s]    = (const float*)d_in[base + 5];
    W2[s]    = (const float*)d_in[base + 6];
    b2[s]    = (const float*)d_in[base + 7];
    gam[s]   = (const float*)d_in[base + 8];
    bet[s]   = (const float*)d_in[base + 9];
  }

  // ---- workspace carve-up (all f16 weight buffers are in B^T [N,K] layout)
  char* ws = (char*)d_ws;
  size_t off = 0;
  auto alloc = [&](size_t bytes) { char* p = ws + off; off = (off + bytes + 255) & ~(size_t)255; return p; };
  _Float16* lstmW[2]; lstmW[0] = (_Float16*)alloc(512L * 256 * 2); lstmW[1] = (_Float16*)alloc(512L * 256 * 2);
  _Float16* sentW = (_Float16*)alloc(128L * 256 * 2);   // [N=128,K=256]
  _Float16* nerW  = (_Float16*)alloc(128L * 64 * 2);    // [N=128,K=64]
  _Float16 *fcW[2], *W1t[2], *W2t[2];
  float* attE[2];
  for (int s = 0; s < 2; ++s) {
    fcW[s] = (_Float16*)alloc(128L * 128 * 2);          // [N=128,K=128]
    W1t[s] = (_Float16*)alloc(512L * 128 * 2);          // [N=512,K=128]
    W2t[s] = (_Float16*)alloc(128L * 512 * 2);          // [N=128,K=512]
    attE[s] = (float*)alloc(50L * 4 * 4);
  }
  _Float16* hs_cat = (_Float16*)alloc((long)BT * 256 * 2);
  float* sent  = (float*)alloc((long)NS * 128 * 4);
  float* nerS  = (float*)alloc((long)NS * 128 * 4);
  float* word  = (float*)alloc((long)NW * 128 * 4);
  float* zbuf  = (float*)alloc((long)NW * 128 * 4);
  float* attS  = (float*)alloc((long)NW * 4 * 4);
  float* ebuf  = (float*)alloc(NE * 4 * 4);
  unsigned* mbuf = (unsigned*)alloc((long)NW * 4 * 4);
  float* den   = (float*)alloc((long)NW * 4 * 4);
  float* acc   = (float*)alloc((long)NW * 128 * 4);

  // ---- 1. weight conversions + attention-edge tables
  long nwK = 512L * 128;
  // LSTM Wt[n][0:128]=Wih[n,:], Wt[n][128:256]=Whh[n,:]  (rows = gate units)
  k_cvt_stride<<<cdivl(nwK, 256), 256, 0, stream>>>(lstmW[0], Wih_f, 512, 128, 256, 0);
  k_cvt_stride<<<cdivl(nwK, 256), 256, 0, stream>>>(lstmW[0], Whh_f, 512, 128, 256, 128);
  k_cvt_stride<<<cdivl(nwK, 256), 256, 0, stream>>>(lstmW[1], Wih_b, 512, 128, 256, 0);
  k_cvt_stride<<<cdivl(nwK, 256), 256, 0, stream>>>(lstmW[1], Whh_b, 512, 128, 256, 128);
  k_cvt_t<<<cdivl(256L * 128, 256), 256, 0, stream>>>(sentW, W_sent_proj, 256, 128);
  k_cvt_t<<<cdivl(64L * 128, 256), 256, 0, stream>>>(nerW, W_ner_proj, 64, 128);
  for (int s = 0; s < 2; ++s) {
    k_cvt_t<<<cdivl(128L * 128, 256), 256, 0, stream>>>(fcW[s], Wfc[s], 128, 128);
    k_cvt_t<<<cdivl(128L * 512, 256), 256, 0, stream>>>(W1t[s], W1[s], 128, 512);
    k_cvt_t<<<cdivl(512L * 128, 256), 256, 0, stream>>>(W2t[s], W2[s], 512, 128);
    k_att_edge<<<1, 256, 0, stream>>>(attE[s], ws_table, Wfeat[s], bfeat[s], Wattn[s]);
  }

  // ---- 2. node feature init
  k_gather_rows<<<cdivl((long)NW * 128, 256), 256, 0, stream>>>(word, word_emb, wid, NW, 128);
  k_gemm<float><<<dim3(cdivl(NS, 16), 1), 256, 0, stream>>>(ner_emb, ner_id, nerW, nullptr, nerS, NS, 128, 64);

  // ---- 3. BiLSTM + sentence projection
  k_lstm<<<dim3(2, 2), 256, 0, stream>>>(local_feature, conv_mask, lstmW[0], lstmW[1],
                                         bih_f, bhh_f, bih_b, bhh_b, hs_cat);
  k_gemm<_Float16><<<dim3(cdivl(NS, 16), 1), 256, 0, stream>>>(hs_cat, utter_idx, sentW, nullptr, sent, NS, 128, 256);

  // ---- 4. wswgat passes
  auto wswgat = [&](const float* hsrc, int Nsrc, const float* origin, float* dest, int n_dst,
                    const int* gidx, const int* sidx, int p) {
    k_gemm<float><<<dim3(cdivl(Nsrc, 16), 1), 256, 0, stream>>>(hsrc, nullptr, fcW[p], nullptr, zbuf, Nsrc, 128, 128);
    k_att_src<<<cdivl((long)Nsrc * 4, 256), 256, 0, stream>>>(attS, zbuf, Wattn[p], Nsrc);
    k_fill_u32<<<cdivl((long)n_dst * 4, 256), 256, 0, stream>>>(mbuf, 0u, (long)n_dst * 4);
    k_fill_u32<<<cdivl((long)n_dst * 4, 256), 256, 0, stream>>>((unsigned*)den, 0u, (long)n_dst * 4);
    k_fill_u32<<<cdivl((long)n_dst * 128, 256), 256, 0, stream>>>((unsigned*)acc, 0u, (long)n_dst * 128);
    k_edge_logits<<<cdivl(NE * 4, 256), 256, 0, stream>>>(ebuf, mbuf, attS, attE[p], gidx, sidx, ws_link, NE);
    k_edge_expsum<<<cdivl(NE * 4, 256), 256, 0, stream>>>(ebuf, mbuf, den, sidx, NE);
    k_edge_scatter<<<cdivl(NE * 128, 256), 256, 0, stream>>>(acc, ebuf, den, zbuf, gidx, sidx, NE);
    k_post_ffn<<<cdivl(n_dst, 16), 128, 0, stream>>>(acc, origin, dest, W1t[p], b1[p], W2t[p], b2[p],
                                                     gam[p], bet[p], n_dst);
  };

  float* out = (float*)d_out;
  // A: word = wswgat(sent, word, s2w)   src gathered by edge_dst, scattered to edge_src
  wswgat(sent, NS, word, word, NW, edge_dst, edge_src, 1);
  // B: ner = wswgat(word, ner, w2s)
  wswgat(word, NW, nerS, nerS, NS, edge_src, edge_dst, 0);
  // C: word = wswgat(ner, word, s2w)
  wswgat(nerS, NS, word, word, NW, edge_dst, edge_src, 1);
  // D: sent = wswgat(word, sent, w2s)
  wswgat(word, NW, sent, sent, NS, edge_src, edge_dst, 0);
  // E: word = wswgat(sent, word, s2w)  -> final output
  wswgat(sent, NS, word, out, NW, edge_dst, edge_src, 1);
}
